// ScaledDotProductAttention_48095043780628
// MI455X (gfx1250) — matrix operand order repaired and checked by reading, not compiled, not run on hardware
//
#include <hip/hip_runtime.h>
#include <hip/hip_bf16.h>

// ---------------------------------------------------------------------------
// ScaledDotProductAttention for MI455X (gfx1250), wave32 + WMMA bf16 +
// async global->LDS staging.
// B=4 S=2048 D=768 H=12 DH=64.  Outputs: ctx [B,S,D] f32 then probs [B,H,S,S] f32.
// ---------------------------------------------------------------------------

#define NB  4
#define NS  2048
#define ND  768
#define NH  12
#define NDH 64

typedef __attribute__((ext_vector_type(16))) __bf16 v16bf;
typedef __attribute__((ext_vector_type(8)))  __bf16 bf16x8;
typedef __attribute__((ext_vector_type(8)))  float  v8f;

static constexpr size_t QKV_ELEMS = (size_t)NB * NH * NS * NDH;   // 6,291,456
static constexpr size_t CTX_ELEMS = (size_t)NB * NS * ND;         // 6,291,456
static constexpr size_t W_ELEMS   = (size_t)ND * ND;              // 589,824

__device__ __forceinline__ v8f wmma_bf16(v16bf a, v16bf b, v8f c) {
  // (neg_a, A, neg_b, B, c_mod, C, reuse_a, reuse_b)
  return __builtin_amdgcn_wmma_f32_16x16x32_bf16(false, a, false, b, (short)0, c,
                                                 false, false);
}

__device__ __forceinline__ v16bf cat8(bf16x8 lo, bf16x8 hi) {
  return __builtin_shufflevector(lo, hi, 0, 1, 2, 3, 4, 5, 6, 7,
                                 8, 9, 10, 11, 12, 13, 14, 15);
}

// A-fragment (16x32, 16-bit layout): lane m holds K = hi*16+e (e<16).
// Source is f32, converted to bf16; explicit float4 loads.
__device__ __forceinline__ v16bf load_a_from_f32(const float* __restrict__ p, int hi) {
  const float4 x0 = *(const float4*)(p + hi * 16);
  const float4 x1 = *(const float4*)(p + hi * 16 + 4);
  const float4 x2 = *(const float4*)(p + hi * 16 + 8);
  const float4 x3 = *(const float4*)(p + hi * 16 + 12);
  v16bf a;
  a[0]  = (__bf16)x0.x; a[1]  = (__bf16)x0.y; a[2]  = (__bf16)x0.z; a[3]  = (__bf16)x0.w;
  a[4]  = (__bf16)x1.x; a[5]  = (__bf16)x1.y; a[6]  = (__bf16)x1.z; a[7]  = (__bf16)x1.w;
  a[8]  = (__bf16)x2.x; a[9]  = (__bf16)x2.y; a[10] = (__bf16)x2.z; a[11] = (__bf16)x2.w;
  a[12] = (__bf16)x3.x; a[13] = (__bf16)x3.y; a[14] = (__bf16)x3.z; a[15] = (__bf16)x3.w;
  return a;
}

// ---------------------------------------------------------------------------
// Kernel 0: Wt[n][k] = bf16(W[k][n]).  32x32 LDS tile transpose.
// Makes the projection GEMM's B-fragments contiguous b128 loads.
// ---------------------------------------------------------------------------
__global__ void w_transpose_bf16_kernel(const float* __restrict__ W,
                                        __bf16* __restrict__ Wt) {
  __shared__ float tile[32][33];
  const int tx = threadIdx.x;          // 0..31
  const int ty = threadIdx.y;          // 0..7
  const int k0 = blockIdx.x * 32;
  const int n0 = blockIdx.y * 32;
  for (int r = ty; r < 32; r += 8)
    tile[r][tx] = W[(size_t)(k0 + r) * ND + n0 + tx];
  __syncthreads();
  for (int r = ty; r < 32; r += 8)
    Wt[(size_t)(n0 + r) * ND + k0 + tx] = (__bf16)tile[tx][r];
}

// ---------------------------------------------------------------------------
// Kernel 1: projection GEMM  out[b][h][s][d] = (x @ W + bias), stored bf16.
// x:[B*S, D] f32, Wt:[out=D][in=D] bf16 (pre-transposed), bias:[D] f32.
// Block: 256 threads (8 waves). Wave w: rows row0..row0+15, cols h*64..h*64+63.
// ---------------------------------------------------------------------------
__global__ void proj_gemm_kernel(const float* __restrict__ x,
                                 const __bf16* __restrict__ Wt,
                                 const float* __restrict__ bias,
                                 __bf16* __restrict__ out) {
  const int lane = threadIdx.x & 31;
  const int wave = threadIdx.x >> 5;
  const int m    = lane & 15;
  const int hi   = lane >> 4;
  const int row0 = blockIdx.x * 128 + wave * 16;  // over B*S = 8192 rows
  const int h    = blockIdx.y;                    // 0..11  (n0 = h*64)
  const int n0   = h * NDH;

  v8f acc[4] = {};

  for (int k0 = 0; k0 < ND; k0 += 32) {
    v16bf a = load_a_from_f32(x + (size_t)(row0 + m) * ND + k0, hi);
#pragma unroll
    for (int t = 0; t < 4; ++t) {
      // B fragment: element e <-> K = k0+hi*16+e, N = n0+t*16+m; contiguous in Wt.
      const __bf16* wr = Wt + (size_t)(n0 + t * 16 + m) * ND + k0 + hi * 16;
      v16bf bf = cat8(*(const bf16x8*)wr, *(const bf16x8*)(wr + 8));
      acc[t] = wmma_bf16(a, bf, acc[t]);
    }
  }

  // Epilogue: +bias, store bf16 in [B][H][S][DH] layout.
#pragma unroll
  for (int t = 0; t < 4; ++t) {
    const int d    = t * 16 + m;
    const float bb = bias[n0 + d];
#pragma unroll
    for (int r = 0; r < 8; ++r) {
      const int row = row0 + r + 8 * hi;          // C layout: M = r + 8*hi
      const int b   = row >> 11;                  // /2048
      const int s   = row & (NS - 1);
      out[(((size_t)b * NH + h) * NS + s) * NDH + d] = (__bf16)(acc[t][r] + bb);
    }
  }
}

// ---------------------------------------------------------------------------
// Kernel 2: scores = (Q K^T)/8, mask -> -1e9, row softmax -> probs (f32).
// One wave (32 threads) per 16-row Q tile; 128KB dynamic LDS holds 16x2048 f32.
// ---------------------------------------------------------------------------
__global__ void scores_softmax_kernel(const __bf16* __restrict__ Qh,
                                      const __bf16* __restrict__ Kh,
                                      const unsigned char* __restrict__ mask,
                                      float* __restrict__ probs) {
  extern __shared__ float srow[];  // [16][2048]
  const int lane = threadIdx.x & 31;
  const int m    = lane & 15;
  const int hi   = lane >> 4;
  const int q0   = blockIdx.x * 16;
  const int bh   = blockIdx.y;     // b*H + h
  const int b    = bh / NH;

  const __bf16* qr = Qh + ((size_t)bh * NS + q0 + m) * NDH;
  const __bf16* Kb = Kh + (size_t)bh * NS * NDH;

  // A fragments (Q tile 16x64 -> two 16x32 bf16 fragments), loaded once.
  v16bf a0 = cat8(*(const bf16x8*)(qr + hi * 16),
                  *(const bf16x8*)(qr + hi * 16 + 8));
  v16bf a1 = cat8(*(const bf16x8*)(qr + 32 + hi * 16),
                  *(const bf16x8*)(qr + 32 + hi * 16 + 8));

  for (int n0 = 0; n0 < NS; n0 += 16) {
    const __bf16* kr = Kb + (size_t)(n0 + m) * NDH;   // K row = score column
    if (n0 + 16 < NS) __builtin_prefetch(kr + 16 * NDH, 0, 1);
    v16bf b0 = cat8(*(const bf16x8*)(kr + hi * 16),
                    *(const bf16x8*)(kr + hi * 16 + 8));
    v16bf b1 = cat8(*(const bf16x8*)(kr + 32 + hi * 16),
                    *(const bf16x8*)(kr + 32 + hi * 16 + 8));
    v8f acc = {};
    acc = wmma_bf16(a0, b0, acc);
    acc = wmma_bf16(a1, b1, acc);

    const unsigned char* mb = mask + ((size_t)b * NS + q0) * NS + n0 + m;
#pragma unroll
    for (int r = 0; r < 8; ++r) {
      const int M = r + 8 * hi;
      float sc = acc[r] * 0.125f;                 // 1/sqrt(64)
      if (mb[(size_t)M * NS]) sc = -1.0e9f;
      srow[M * NS + n0 + m] = sc;
    }
  }

  // Row softmax: whole wave cooperates on each of the 16 rows.
  for (int r = 0; r < 16; ++r) {
    float mx = -3.0e38f;
    for (int c = lane; c < NS; c += 32) mx = fmaxf(mx, srow[r * NS + c]);
#pragma unroll
    for (int off = 16; off > 0; off >>= 1) mx = fmaxf(mx, __shfl_xor(mx, off, 32));
    float sum = 0.0f;
    for (int c = lane; c < NS; c += 32) {
      float e = __expf(srow[r * NS + c] - mx);
      srow[r * NS + c] = e;
      sum += e;
    }
#pragma unroll
    for (int off = 16; off > 0; off >>= 1) sum += __shfl_xor(sum, off, 32);
    const float inv = 1.0f / sum;
    float* pr = probs + ((size_t)bh * NS + q0 + r) * NS;
    for (int c = lane; c < NS; c += 32) pr[c] = srow[r * NS + c] * inv;  // coalesced
  }
}

// ---------------------------------------------------------------------------
// Kernel 3: ctx = probs @ V.  256 threads (8 waves) share a 32x64 V tile that
// is staged with GLOBAL_LOAD_ASYNC_TO_LDS_B128 (ASYNCcnt-tracked, no VGPRs).
// probs f32 -> bf16 A fragments on the fly; output f32 [B][S][H*DH].
// ---------------------------------------------------------------------------
__global__ void ctx_gemm_kernel(const float* __restrict__ probs,
                                const __bf16* __restrict__ Vh,
                                float* __restrict__ ctx) {
  __shared__ __align__(16) __bf16 vt[32 * NDH];   // 32x64 V tile, 4KB
  const int lane = threadIdx.x & 31;
  const int wave = threadIdx.x >> 5;
  const int m    = lane & 15;
  const int hi   = lane >> 4;
  const int bh   = blockIdx.y;
  const int b    = bh / NH;
  const int h    = bh % NH;
  const int q0   = blockIdx.x * 128 + wave * 16;

  const float*  pbase = probs + ((size_t)bh * NS + q0 + m) * NS;
  const __bf16* Vb    = Vh + (size_t)bh * NS * NDH;

  // Per-lane LDS byte offset for this thread's 16B chunk of the V tile.
  // Generic (flat) shared pointers carry the LDS offset in their low 32 bits.
  const unsigned lds_off =
      (unsigned)(size_t)(&vt[0]) + (unsigned)threadIdx.x * 16u;

  v8f acc[4] = {};

  for (int k0 = 0; k0 < NS; k0 += 32) {
    // Async stage: 256 lanes x 16B = 4KB tile, memory -> LDS, ASYNCcnt-tracked.
    const unsigned long long gaddr =
        (unsigned long long)(const void*)(Vb + (size_t)k0 * NDH) +
        (unsigned long long)threadIdx.x * 16ull;
    asm volatile("global_load_async_to_lds_b128 %0, %1, off"
                 :: "v"(lds_off), "v"(gaddr)
                 : "memory");
    asm volatile("s_wait_asynccnt 0x0" ::: "memory");
    __syncthreads();

    v16bf a = load_a_from_f32(pbase + k0, hi);
#pragma unroll
    for (int t = 0; t < 4; ++t) {
      v16bf bf;
#pragma unroll
      for (int e = 0; e < 16; ++e)
        bf[e] = vt[(hi * 16 + e) * NDH + t * 16 + m];
      acc[t] = wmma_bf16(a, bf, acc[t]);
    }
    __syncthreads();
  }

#pragma unroll
  for (int t = 0; t < 4; ++t) {
    const int d = t * 16 + m;
#pragma unroll
    for (int r = 0; r < 8; ++r) {
      const int q = q0 + r + 8 * hi;
      ctx[((size_t)(b * NS + q)) * ND + h * NDH + d] = acc[t][r];
    }
  }
}

// ---------------------------------------------------------------------------
extern "C" void kernel_launch(void* const* d_in, const int* in_sizes, int n_in,
                              void* d_out, int out_size, void* d_ws, size_t ws_size,
                              hipStream_t stream) {
  const float* q  = (const float*)d_in[0];
  const float* k  = (const float*)d_in[1];
  const float* v  = (const float*)d_in[2];
  const unsigned char* mask = (const unsigned char*)d_in[3];  // bool mask, 1 = masked out
  const float* Wq = (const float*)d_in[4];
  const float* bq = (const float*)d_in[5];
  const float* Wk = (const float*)d_in[6];
  const float* bk = (const float*)d_in[7];
  const float* Wv = (const float*)d_in[8];
  const float* bv = (const float*)d_in[9];

  // Workspace: Q/K/V heads bf16 [B][H][S][DH] + transposed bf16 weights.
  __bf16* Qh  = (__bf16*)d_ws;
  __bf16* Kh  = Qh + QKV_ELEMS;
  __bf16* Vh  = Kh + QKV_ELEMS;
  __bf16* WqT = Vh + QKV_ELEMS;
  __bf16* WkT = WqT + W_ELEMS;
  __bf16* WvT = WkT + W_ELEMS;

  float* ctx   = (float*)d_out;           // [B,S,D]
  float* probs = ctx + CTX_ELEMS;         // [B,H,S,S]

  dim3 gt(ND / 32, ND / 32);              // 24 x 24 blocks
  dim3 bt(32, 8);
  w_transpose_bf16_kernel<<<gt, bt, 0, stream>>>(Wq, WqT);
  w_transpose_bf16_kernel<<<gt, bt, 0, stream>>>(Wk, WkT);
  w_transpose_bf16_kernel<<<gt, bt, 0, stream>>>(Wv, WvT);

  dim3 gp(NB * NS / 128, NH);             // 64 x 12 blocks, 256 thr
  proj_gemm_kernel<<<gp, 256, 0, stream>>>(q, WqT, bq, Qh);
  proj_gemm_kernel<<<gp, 256, 0, stream>>>(k, WkT, bk, Kh);
  proj_gemm_kernel<<<gp, 256, 0, stream>>>(v, WvT, bv, Vh);

  dim3 gs(NS / 16, NB * NH);              // 128 x 48 blocks, 1 wave each
  scores_softmax_kernel<<<gs, 32, 16 * NS * sizeof(float), stream>>>(Qh, Kh, mask, probs);

  dim3 gc(NS / 128, NB * NH);             // 16 x 48 blocks, 256 thr
  ctx_gemm_kernel<<<gc, 256, 0, stream>>>(probs, Vh, ctx);
}
